// BatchDenseGAT_35562329211432
// MI455X (gfx1250) — matrix operand order repaired and checked by reading, not compiled, not run on hardware
//
#include <hip/hip_runtime.h>
#include <hip/hip_bf16.h>

typedef __attribute__((ext_vector_type(16))) __bf16 v16bf;
typedef __attribute__((ext_vector_type(8)))  float  v8f;

#define BS  32
#define N   512
#define DH  160   // layer-1 input feats (64+64+32)
#define NH1 8
#define O1  64
#define F2  512   // layer-2 input feats (8*64)
#define O2  16

static __device__ __forceinline__ float lrelu(float x) { return x >= 0.f ? x : 0.2f * x; }

// ---------- k0: h = concat(emb0[v], emb1[v], local) -> bf16 [BS,N,160] ----------
__global__ void k_prep_h(const int* __restrict__ verts,
                         const float* __restrict__ local_emb,
                         const float* __restrict__ emb0,
                         const float* __restrict__ emb1,
                         __bf16* __restrict__ h)
{
  int i = blockIdx.x * 256 + threadIdx.x;
  const int total = BS * N * DH;
  if (i >= total) return;
  int f = i % DH, bn = i / DH;
  int v = verts[bn];
  float x;
  if (f < 64)        x = emb0[(size_t)v * 64 + f];
  else if (f < 128)  x = emb1[(size_t)v * 64 + (f - 64)];
  else               x = local_emb[(size_t)bn * 32 + (f - 128)];
  h[i] = (__bf16)x;
}

// ---------- k1: transpose weights to K-contiguous bf16 ----------
// w1 [8,160,64] -> w1T [8,64,160] ; w2 [512,16] -> w2T [16,512]
__global__ void k_prep_w(const float* __restrict__ w1, const float* __restrict__ w2,
                         __bf16* __restrict__ w1T, __bf16* __restrict__ w2T)
{
  int i = blockIdx.x * 256 + threadIdx.x;
  const int n1 = NH1 * O1 * DH;
  if (i < n1) {
    int head = i / (O1 * DH); int rem = i % (O1 * DH);
    int o = rem / DH; int f = rem % DH;
    w1T[i] = (__bf16)w1[((size_t)head * DH + f) * O1 + o];
  } else if (i < n1 + O2 * F2) {
    int j = i - n1;
    int o = j / F2; int f = j % F2;
    w2T[j] = (__bf16)w2[(size_t)f * O2 + o];
  }
}

// ---------- k2: pack adjacency to bitmask [BS,N,16] u32 ----------
__global__ void k_pack_adj(const float* __restrict__ adj, unsigned int* __restrict__ bits)
{
  int i = blockIdx.x * 256 + threadIdx.x;
  if (i >= BS * N * (N / 32)) return;
  const float* src = adj + (size_t)i * 32;
  unsigned int w = 0;
#pragma unroll
  for (int b = 0; b < 32; ++b) if (src[b] > 0.f) w |= (1u << b);
  bits[i] = w;
}

// ---------- k3: hp1 = h @ w1 (per b,head), store hp1T bf16, fuse s/d ----------
__global__ void __launch_bounds__(32)
k_gemm1(const __bf16* __restrict__ h, const __bf16* __restrict__ w1T,
        const float* __restrict__ a_src1, const float* __restrict__ a_dst1,
        __bf16* __restrict__ hp1T, float* __restrict__ s1, float* __restrict__ d1)
{
  const int bh = blockIdx.x;            // b*8+head
  const int head = bh & 7;
  const int b = bh >> 3;
  const int n0 = blockIdx.y * 16;
  const int lane = threadIdx.x;
  const int half = lane >> 4, lq = lane & 15;

  v8f c[4] = {};
  const __bf16* arow = h + ((size_t)(b * N + n0 + lq)) * DH + half * 16;
  for (int ks = 0; ks < 5; ++ks) {
    const int k0 = ks * 32;
    v16bf A;
#pragma unroll
    for (int e = 0; e < 8; ++e) {
      A[e]     = arow[k0 + e];          // K = k0 + e      (+16 for upper half)
      A[e + 8] = arow[k0 + 8 + e];      // K = k0 + 8 + e  (+16 for upper half)
    }
#pragma unroll
    for (int ot = 0; ot < 4; ++ot) {
      const int o = ot * 16 + lq;
      const v16bf Bv = *(const v16bf*)(w1T + ((size_t)(head * O1 + o)) * DH + k0 + half * 16);
      c[ot] = __builtin_amdgcn_wmma_f32_16x16x32_bf16(false, A, false, Bv,
                                                      (short)0, c[ot], false, false);
    }
  }
  float aS[4], aD[4];
#pragma unroll
  for (int ot = 0; ot < 4; ++ot) {
    aS[ot] = a_src1[head * O1 + ot * 16 + lq];
    aD[ot] = a_dst1[head * O1 + ot * 16 + lq];
  }
#pragma unroll
  for (int r = 0; r < 8; ++r) {
    const int m = n0 + half * 8 + r;    // C row M = r + 8*half
    float sv = 0.f, dv = 0.f;
#pragma unroll
    for (int ot = 0; ot < 4; ++ot) {
      float val = c[ot][r];
      hp1T[((size_t)bh * O1 + ot * 16 + lq) * N + m] = (__bf16)val; // transposed store
      float t = tanhf(val);
      sv += t * aS[ot]; dv += t * aD[ot];
    }
    for (int off = 8; off >= 1; off >>= 1) {   // reduce across the 16 lanes of this half
      sv += __shfl_xor(sv, off, 32);
      dv += __shfl_xor(dv, off, 32);
    }
    if (lq == 0) { s1[(size_t)bh * N + m] = sv; d1[(size_t)bh * N + m] = dv; }
  }
}

// ---------- k4: layer-1 attention (flash-style, scores in LDS) + ELU -> h2 ----------
__global__ void __launch_bounds__(32)
k_flash1(const float* __restrict__ s1, const float* __restrict__ d1,
         const unsigned int* __restrict__ adjbits,
         const __bf16* __restrict__ hp1T, __bf16* __restrict__ h2)
{
  __shared__ float p[16 * N];
  __shared__ float srow[16];
  __shared__ float rinv_s[16];
  const int bh = blockIdx.x; const int head = bh & 7; const int b = bh >> 3;
  const int n0 = blockIdx.y * 16;
  const int lane = threadIdx.x;
  const int half = lane >> 4, lq = lane & 15;

  if (lane < 16) srow[lane] = s1[(size_t)bh * N + n0 + lane];
  __syncthreads();

  const float* dvec = d1 + (size_t)bh * N;
  const unsigned int* ab = adjbits + ((size_t)b * N + n0) * (N / 32);
  for (int i = lane; i < 16 * N; i += 32) {
    int row = i >> 9, m = i & (N - 1);
    unsigned int w = ab[row * (N / 32) + (m >> 5)];
    float sc = -3.0e38f;
    if ((w >> (m & 31)) & 1u) sc = lrelu(srow[row] + dvec[m]);
    p[i] = sc;
  }
  __syncthreads();
  {
    const int m0 = half * 256;           // 2 lanes per row, half each
    float mx = -3.0e38f;
    for (int m = 0; m < 256; ++m) mx = fmaxf(mx, p[lq * N + m0 + m]);
    mx = fmaxf(mx, __shfl_xor(mx, 16, 32));
    float sum = 0.f;
    for (int m = 0; m < 256; ++m) {
      float e = __expf(p[lq * N + m0 + m] - mx);
      p[lq * N + m0 + m] = e; sum += e;
    }
    sum += __shfl_xor(sum, 16, 32);
    if (half == 0) rinv_s[lq] = 1.0f / sum;   // self-loop guarantees sum > 0
  }
  __syncthreads();

  const float rinv = rinv_s[lq];
  v8f c[4] = {};
  for (int ks = 0; ks < 16; ++ks) {
    const int k0 = ks * 32;
    v16bf A;
    const float* prow = p + lq * N + k0 + half * 16;
#pragma unroll
    for (int e = 0; e < 8; ++e) {
      A[e]     = (__bf16)(prow[e] * rinv);
      A[e + 8] = (__bf16)(prow[8 + e] * rinv);
    }
#pragma unroll
    for (int ot = 0; ot < 4; ++ot) {
      const int o = ot * 16 + lq;
      const v16bf Bv = *(const v16bf*)(hp1T + ((size_t)bh * O1 + o) * N + k0 + half * 16);
      c[ot] = __builtin_amdgcn_wmma_f32_16x16x32_bf16(false, A, false, Bv,
                                                      (short)0, c[ot], false, false);
    }
  }
#pragma unroll
  for (int ot = 0; ot < 4; ++ot) {
#pragma unroll
    for (int r = 0; r < 8; ++r) {
      const int n = n0 + half * 8 + r;
      const int feat = head * O1 + ot * 16 + lq;   // [b,n,h*64+o] layout == layer-2 input
      float v = c[ot][r];
      v = v > 0.f ? v : (__expf(v) - 1.0f);        // ELU
      h2[((size_t)b * N + n) * F2 + feat] = (__bf16)v;
    }
  }
}

// ---------- k5: hp2 = h2 @ w2, store hp2T bf16, fuse s2/d2 ----------
__global__ void __launch_bounds__(32)
k_gemm2(const __bf16* __restrict__ h2, const __bf16* __restrict__ w2T,
        const float* __restrict__ a_src2, const float* __restrict__ a_dst2,
        __bf16* __restrict__ hp2T, float* __restrict__ s2, float* __restrict__ d2)
{
  const int b = blockIdx.x;
  const int n0 = blockIdx.y * 16;
  const int lane = threadIdx.x;
  const int half = lane >> 4, lq = lane & 15;

  v8f c = {};
  const __bf16* arow = h2 + ((size_t)(b * N + n0 + lq)) * F2 + half * 16;
  for (int ks = 0; ks < 16; ++ks) {
    const int k0 = ks * 32;
    v16bf A;
#pragma unroll
    for (int e = 0; e < 8; ++e) { A[e] = arow[k0 + e]; A[e + 8] = arow[k0 + 8 + e]; }
    const v16bf Bv = *(const v16bf*)(w2T + (size_t)lq * F2 + k0 + half * 16);
    c = __builtin_amdgcn_wmma_f32_16x16x32_bf16(false, A, false, Bv,
                                                (short)0, c, false, false);
  }
  const float aS = a_src2[lq], aD = a_dst2[lq];
#pragma unroll
  for (int r = 0; r < 8; ++r) {
    const int m = n0 + half * 8 + r;
    float val = c[r];
    hp2T[((size_t)b * O2 + lq) * N + m] = (__bf16)val;
    float t = tanhf(val);
    float sv = t * aS, dv = t * aD;
    for (int off = 8; off >= 1; off >>= 1) {
      sv += __shfl_xor(sv, off, 32);
      dv += __shfl_xor(dv, off, 32);
    }
    if (lq == 0) { s2[(size_t)b * N + m] = sv; d2[(size_t)b * N + m] = dv; }
  }
}

// ---------- k6: layer-2 attention -> d_out (f32) ----------
__global__ void __launch_bounds__(32)
k_flash2(const float* __restrict__ s2, const float* __restrict__ d2,
         const unsigned int* __restrict__ adjbits,
         const __bf16* __restrict__ hp2T, float* __restrict__ out)
{
  __shared__ float p[16 * N];
  __shared__ float srow[16];
  __shared__ float rinv_s[16];
  const int b = blockIdx.x;
  const int n0 = blockIdx.y * 16;
  const int lane = threadIdx.x;
  const int half = lane >> 4, lq = lane & 15;

  if (lane < 16) srow[lane] = s2[(size_t)b * N + n0 + lane];
  __syncthreads();

  const float* dvec = d2 + (size_t)b * N;
  const unsigned int* ab = adjbits + ((size_t)b * N + n0) * (N / 32);
  for (int i = lane; i < 16 * N; i += 32) {
    int row = i >> 9, m = i & (N - 1);
    unsigned int w = ab[row * (N / 32) + (m >> 5)];
    float sc = -3.0e38f;
    if ((w >> (m & 31)) & 1u) sc = lrelu(srow[row] + dvec[m]);
    p[i] = sc;
  }
  __syncthreads();
  {
    const int m0 = half * 256;
    float mx = -3.0e38f;
    for (int m = 0; m < 256; ++m) mx = fmaxf(mx, p[lq * N + m0 + m]);
    mx = fmaxf(mx, __shfl_xor(mx, 16, 32));
    float sum = 0.f;
    for (int m = 0; m < 256; ++m) {
      float e = __expf(p[lq * N + m0 + m] - mx);
      p[lq * N + m0 + m] = e; sum += e;
    }
    sum += __shfl_xor(sum, 16, 32);
    if (half == 0) rinv_s[lq] = 1.0f / sum;
  }
  __syncthreads();

  const float rinv = rinv_s[lq];
  v8f c = {};
  for (int ks = 0; ks < 16; ++ks) {
    const int k0 = ks * 32;
    v16bf A;
    const float* prow = p + lq * N + k0 + half * 16;
#pragma unroll
    for (int e = 0; e < 8; ++e) {
      A[e]     = (__bf16)(prow[e] * rinv);
      A[e + 8] = (__bf16)(prow[8 + e] * rinv);
    }
    const v16bf Bv = *(const v16bf*)(hp2T + ((size_t)b * O2 + lq) * N + k0 + half * 16);
    c = __builtin_amdgcn_wmma_f32_16x16x32_bf16(false, A, false, Bv,
                                                (short)0, c, false, false);
  }
#pragma unroll
  for (int r = 0; r < 8; ++r) {
    const int n = n0 + half * 8 + r;
    out[((size_t)b * N + n) * O2 + lq] = c[r];   // mean over 1 head == identity
  }
}

// ---------- k7: in-place log_softmax over 16 classes ----------
__global__ void k_logsm(float* __restrict__ out)
{
  int row = blockIdx.x * 256 + threadIdx.x;
  if (row >= BS * N) return;
  float* pr = out + (size_t)row * O2;
  float v[O2];
  float mx = -3.0e38f;
#pragma unroll
  for (int i = 0; i < O2; ++i) { v[i] = pr[i]; mx = fmaxf(mx, v[i]); }
  float s = 0.f;
#pragma unroll
  for (int i = 0; i < O2; ++i) s += __expf(v[i] - mx);
  float lse = mx + __logf(s);
#pragma unroll
  for (int i = 0; i < O2; ++i) pr[i] = v[i] - lse;
}

extern "C" void kernel_launch(void* const* d_in, const int* in_sizes, int n_in,
                              void* d_out, int out_size, void* d_ws, size_t ws_size,
                              hipStream_t stream) {
  (void)in_sizes; (void)n_in; (void)out_size; (void)ws_size;
  const float* adj       = (const float*)d_in[0];
  const int*   verts     = (const int*)d_in[1];
  const float* local_emb = (const float*)d_in[2];
  const float* emb0      = (const float*)d_in[3];
  const float* emb1      = (const float*)d_in[4];
  const float* w1        = (const float*)d_in[5];
  const float* a_src1    = (const float*)d_in[6];
  const float* a_dst1    = (const float*)d_in[7];
  const float* w2        = (const float*)d_in[8];
  const float* a_src2    = (const float*)d_in[9];
  const float* a_dst2    = (const float*)d_in[10];
  float* out = (float*)d_out;

  char* ws = (char*)d_ws;
  size_t off = 0;
  auto alloc = [&](size_t bytes) -> char* {
    char* p = ws + off; off += (bytes + 255) & ~(size_t)255; return p;
  };
  __bf16* h_b    = (__bf16*)alloc((size_t)BS * N * DH * 2);          // 5.2 MB
  __bf16* w1T    = (__bf16*)alloc((size_t)NH1 * O1 * DH * 2);
  __bf16* w2T    = (__bf16*)alloc((size_t)O2 * F2 * 2);
  unsigned int* adjb = (unsigned int*)alloc((size_t)BS * N * (N / 32) * 4); // 1 MB
  __bf16* hp1T   = (__bf16*)alloc((size_t)BS * NH1 * O1 * N * 2);    // 16.8 MB
  float*  s1     = (float*)alloc((size_t)BS * NH1 * N * 4);
  float*  d1v    = (float*)alloc((size_t)BS * NH1 * N * 4);
  __bf16* h2     = (__bf16*)alloc((size_t)BS * N * F2 * 2);          // 16.8 MB
  __bf16* hp2T   = (__bf16*)alloc((size_t)BS * O2 * N * 2);
  float*  s2     = (float*)alloc((size_t)BS * N * 4);
  float*  d2v    = (float*)alloc((size_t)BS * N * 4);

  k_prep_h<<<(BS * N * DH + 255) / 256, 256, 0, stream>>>(verts, local_emb, emb0, emb1, h_b);
  k_prep_w<<<(NH1 * O1 * DH + O2 * F2 + 255) / 256, 256, 0, stream>>>(w1, w2, w1T, w2T);
  k_pack_adj<<<(BS * N * (N / 32) + 255) / 256, 256, 0, stream>>>(adj, adjb);
  k_gemm1<<<dim3(BS * NH1, N / 16), 32, 0, stream>>>(h_b, w1T, a_src1, a_dst1, hp1T, s1, d1v);
  k_flash1<<<dim3(BS * NH1, N / 16), 32, 0, stream>>>(s1, d1v, adjb, hp1T, h2);
  k_gemm2<<<dim3(BS, N / 16), 32, 0, stream>>>(h2, w2T, a_src2, a_dst2, hp2T, s2, d2v);
  k_flash2<<<dim3(BS, N / 16), 32, 0, stream>>>(s2, d2v, adjb, hp2T, out);
  k_logsm<<<(BS * N + 255) / 256, 256, 0, stream>>>(out);
}
